// Attention_46823733461345
// MI455X (gfx1250) — matrix operand order repaired and checked by reading, not compiled, not run on hardware
//
#include <hip/hip_runtime.h>
#include <hip/hip_bf16.h>
#include <stdint.h>

// ---------------------------------------------------------------------------
// Additive (Bahdanau) attention for MI455X / gfx1250.
//   BS=64, M=2048, D=1024, H=1024
// Dominant cost: m_proj = memory @ Wm^T (274 GFLOP) -> v_wmma_f32_16x16x32_bf16.
// HBM: two 512MB fp32 passes over `memory` ~= 45us at 23.3 TB/s.
// WMMA pipe: ~16.8M wmma ops / 1024 SIMDs ~= 26us. 4(M)x2(N) register tile
// -> 1 ds_b128 + 0.5 global_b128 per WMMA. Second LDS base register forced
// via opaque offset so all DS offsets are immediates (no per-step address VALU).
// ---------------------------------------------------------------------------

#define BSZ   64
#define MDIM  2048
#define DDIM  1024
#define HDIM  1024
#define ROWS  64               // memory rows per score block
#define STRA  (DDIM + 8)       // padded bf16 LDS row stride (2064B, 16B aligned)

typedef __bf16 bf16_t;
typedef __attribute__((ext_vector_type(16))) __bf16 v16bf;
typedef __attribute__((ext_vector_type(8)))  float  v8f;

static __device__ __forceinline__ unsigned short f32_to_bf16(float f) {
    unsigned int u = __float_as_uint(f);
    u += 0x7FFFu + ((u >> 16) & 1u);      // round-to-nearest-even
    return (unsigned short)(u >> 16);
}

// Build a 16x32 bf16 fragment from a pre-offset (lane-resolved) pointer.
// A frags: elems 0..7 at +kk, elems 8..15 at +kk+16 (split K-halves).
// B frags: same split K-halves (gap = 16).
static __device__ __forceinline__ v16bf frag2(const unsigned short* p, int kk, int gap) {
    union { v16bf v; uint4 q[2]; } u;
    u.q[0] = *(const uint4*)(p + kk);
    u.q[1] = *(const uint4*)(p + kk + gap);
    return u.v;
}

// ---- kernel 0: Wm fp32 -> bf16 (L2-resident operand for the big GEMM) ------
extern "C" __global__ void __launch_bounds__(256)
k_cvt_wm(const float* __restrict__ Wm, unsigned short* __restrict__ WmBf) {
    int gid = blockIdx.x * 256 + threadIdx.x;          // HDIM*DDIM/4 threads
    float4 f = ((const float4*)Wm)[gid];
    union { unsigned short h[4]; uint2 u; } pk;
    pk.h[0] = f32_to_bf16(f.x); pk.h[1] = f32_to_bf16(f.y);
    pk.h[2] = f32_to_bf16(f.z); pk.h[3] = f32_to_bf16(f.w);
    ((uint2*)WmBf)[gid] = pk.u;
}

// ---- kernel 1: q_proj = query @ Wq^T + bq  (tiny: 67M MACs) ----------------
extern "C" __global__ void __launch_bounds__(256)
k_qproj(const float* __restrict__ query, const float* __restrict__ Wq,
        const float* __restrict__ bq, float* __restrict__ qproj) {
    int gid = blockIdx.x * 256 + threadIdx.x;          // BSZ*HDIM threads
    int b = gid >> 10, h = gid & (HDIM - 1);
    const float4* q4 = (const float4*)(query + (size_t)b * DDIM);
    const float4* w4 = (const float4*)(Wq + (size_t)h * DDIM);
    float acc = 0.f;
#pragma unroll 4
    for (int i = 0; i < DDIM / 4; ++i) {
        float4 a = q4[i], w = w4[i];
        acc += a.x * w.x + a.y * w.y + a.z * w.z + a.w * w.w;
    }
    qproj[gid] = acc + bq[h];
}

// ---- kernel 2: attn[b][m] = sum_h tanh(qproj[b][h] + (mem@Wm^T)[b][m][h]) * v[h]
// One block = 64 rows of one batch, 8 waves. Each wave: 4 passes of a
// 4(Mtile)x2(Ntile) register tile -> 8 f32 accumulators, 8 WMMAs per k-step.
extern "C" __global__ void __launch_bounds__(256)
k_scores(const float* __restrict__ memory, const unsigned short* __restrict__ WmBf,
         const float* __restrict__ qproj, const float* __restrict__ vvec,
         float* __restrict__ attnOut) {
    extern __shared__ char smem[];
    unsigned short* sA   = (unsigned short*)smem;                       // ROWS*STRA bf16
    float*          sPrt = (float*)(smem + (size_t)ROWS * STRA * 2);    // 8 waves * ROWS

    const int b   = blockIdx.x / (MDIM / ROWS);
    const int m0  = (blockIdx.x % (MDIM / ROWS)) * ROWS;
    const int tid = threadIdx.x;

    // stage 64x1024 fp32 rows -> bf16 in LDS (256 KB read, 128 KB LDS)
    const float4* memF4 = (const float4*)(memory + ((size_t)b * MDIM + m0) * DDIM);
    for (int idx = tid; idx < ROWS * DDIM / 4; idx += 256) {
        int row = idx >> 8;            // 256 float4 per row
        int c4  = idx & 255;
        float4 f = memF4[row * 256 + c4];
        union { unsigned short h[4]; uint2 u; } pk;
        pk.h[0] = f32_to_bf16(f.x); pk.h[1] = f32_to_bf16(f.y);
        pk.h[2] = f32_to_bf16(f.z); pk.h[3] = f32_to_bf16(f.w);
        *(uint2*)(sA + row * STRA + c4 * 4) = pk.u;
    }
    __syncthreads();

    const int wave = tid >> 5;
    const int lane = tid & 31;
    const int n    = lane & 15;

    // Hoisted, lane-resolved base pointers. Rows 32/48 sit past the 64KB DS
    // immediate window; an opaque offset forces a second hoisted base VGPR
    // instead of per-iteration address rematerialization.
    const int aoff = (lane < 16) ? 0 : 8;    // A K-half select
    const int boff = (lane < 16) ? 0 : 8;    // B K-half select
    const unsigned short* ap0 = sA + n * STRA + aoff;
    const unsigned short* ap1 = ap0 + 16 * STRA;         // +33024B: imm OK
    int off32 = 32 * STRA;
    asm("" : "+v"(off32));                               // opaque to const-folding
    const unsigned short* ap2 = ap0 + off32;             // own base register
    const unsigned short* ap3 = ap2 + 16 * STRA;         // +33024B off ap2: imm OK

    // Per-lane partial attn: this lane's column contribution to rows
    // [mt*16 + (lane<16 ? e : 8+e)]; cross-lane reduce happens ONCE at the end.
    float wacc[32];
#pragma unroll
    for (int i = 0; i < 32; ++i) wacc[i] = 0.f;

    for (int jp = 0; jp < 4; ++jp) {
        const int hb0 = (wave * 8 + jp * 2) * 16;     // two adjacent n-tiles
        const unsigned short* bp0 = WmBf + (size_t)(hb0 + n) * DDIM + boff;
        const unsigned short* bp1 = bp0 + 16 * DDIM;

        v8f acc[8];                                    // [nt2*4 + mt]
#pragma unroll
        for (int i = 0; i < 8; ++i) acc[i] = (v8f){};

        for (int kk = 0; kk < DDIM; kk += 32) {
            v16bf bf0 = frag2(bp0, kk, 16);
            v16bf bf1 = frag2(bp1, kk, 16);
            v16bf f0  = frag2(ap0, kk, 16);
            v16bf f1  = frag2(ap1, kk, 16);
            v16bf f2  = frag2(ap2, kk, 16);
            v16bf f3  = frag2(ap3, kk, 16);
            acc[0] = __builtin_amdgcn_wmma_f32_16x16x32_bf16(false, f0, false, bf0, (short)0, acc[0], false, false);
            acc[1] = __builtin_amdgcn_wmma_f32_16x16x32_bf16(false, f1, false, bf0, (short)0, acc[1], false, false);
            acc[2] = __builtin_amdgcn_wmma_f32_16x16x32_bf16(false, f2, false, bf0, (short)0, acc[2], false, false);
            acc[3] = __builtin_amdgcn_wmma_f32_16x16x32_bf16(false, f3, false, bf0, (short)0, acc[3], false, false);
            acc[4] = __builtin_amdgcn_wmma_f32_16x16x32_bf16(false, f0, false, bf1, (short)0, acc[4], false, false);
            acc[5] = __builtin_amdgcn_wmma_f32_16x16x32_bf16(false, f1, false, bf1, (short)0, acc[5], false, false);
            acc[6] = __builtin_amdgcn_wmma_f32_16x16x32_bf16(false, f2, false, bf1, (short)0, acc[6], false, false);
            acc[7] = __builtin_amdgcn_wmma_f32_16x16x32_bf16(false, f3, false, bf1, (short)0, acc[7], false, false);
        }

        // Fused epilogue: per-lane tanh(acc + qp)*v accumulated into wacc.
        // C layout: col N = lane&15; rows 0-7 in lanes<16, rows 8-15 in lanes>=16.
#pragma unroll
        for (int nt2 = 0; nt2 < 2; ++nt2) {
            const int hb = hb0 + nt2 * 16;
            const float qv = qproj[(size_t)b * HDIM + hb + n];
            const float vv = vvec[hb + n];
#pragma unroll
            for (int mt = 0; mt < 4; ++mt) {
                v8f a = acc[nt2 * 4 + mt];
#pragma unroll
                for (int e = 0; e < 8; ++e)
                    wacc[mt * 8 + e] += tanhf(a[e] + qv) * vv;
            }
        }
    }

    // One shuffle tree over the 16 N-lanes (128 bpermutes per wave total).
#pragma unroll
    for (int i = 0; i < 32; ++i) {
        float t = wacc[i];
        t += __shfl_xor(t, 8, 16);
        t += __shfl_xor(t, 4, 16);
        t += __shfl_xor(t, 2, 16);
        t += __shfl_xor(t, 1, 16);
        wacc[i] = t;                       // valid on lanes 0 and 16
    }

    // deterministic cross-wave reduction (no float atomics)
    if (n == 0) {
        const int rbase = (lane < 16) ? 0 : 8;
#pragma unroll
        for (int mt = 0; mt < 4; ++mt)
#pragma unroll
            for (int e = 0; e < 8; ++e)
                sPrt[wave * ROWS + mt * 16 + rbase + e] = wacc[mt * 8 + e];
    }
    __syncthreads();
    if (tid < ROWS) {
        float s = 0.f;
#pragma unroll
        for (int w = 0; w < 8; ++w) s += sPrt[w * ROWS + tid];
        attnOut[(size_t)b * MDIM + m0 + tid] = s;
    }
}

// ---- kernel 3: softmax over M per batch ------------------------------------
extern "C" __global__ void __launch_bounds__(256)
k_softmax(const float* __restrict__ attn, float* __restrict__ weights) {
    __shared__ float sv[MDIM];
    __shared__ float sr[256];
    const int b = blockIdx.x, tid = threadIdx.x;
    float lmax = -3.4e38f;
    for (int i = tid; i < MDIM; i += 256) {
        float x = attn[(size_t)b * MDIM + i];
        sv[i] = x; lmax = fmaxf(lmax, x);
    }
    sr[tid] = lmax; __syncthreads();
    for (int s = 128; s > 0; s >>= 1) {
        if (tid < s) sr[tid] = fmaxf(sr[tid], sr[tid + s]);
        __syncthreads();
    }
    const float mx = sr[0]; __syncthreads();
    float lsum = 0.f;
    for (int i = tid; i < MDIM; i += 256) {
        float e = __expf(sv[i] - mx);
        sv[i] = e; lsum += e;
    }
    sr[tid] = lsum; __syncthreads();
    for (int s = 128; s > 0; s >>= 1) {
        if (tid < s) sr[tid] += sr[tid + s];
        __syncthreads();
    }
    const float inv = 1.0f / sr[0];
    for (int i = tid; i < MDIM; i += 256)
        weights[(size_t)b * MDIM + i] = sv[i] * inv;
}

// ---- kernel 4: weighted_memory[b][d] = sum_m w[b][m] * memory[b][m][d] -----
extern "C" __global__ void __launch_bounds__(256)
k_wmem(const float* __restrict__ weights, const float* __restrict__ memory,
       float* __restrict__ out) {
    __shared__ float sw[MDIM];
    const int b  = blockIdx.x / (DDIM / 256);
    const int d0 = (blockIdx.x % (DDIM / 256)) * 256;
    const int tid = threadIdx.x;
    for (int i = tid; i < MDIM; i += 256) sw[i] = weights[(size_t)b * MDIM + i];
    __syncthreads();
    const float* mp = memory + (size_t)b * MDIM * DDIM + d0 + tid;
    float acc = 0.f;
#pragma unroll 4
    for (int m = 0; m < MDIM; ++m) acc += sw[m] * mp[(size_t)m * DDIM];
    out[(size_t)b * DDIM + d0 + tid] = acc;
}

// ---------------------------------------------------------------------------
extern "C" void kernel_launch(void* const* d_in, const int* in_sizes, int n_in,
                              void* d_out, int out_size, void* d_ws, size_t ws_size,
                              hipStream_t stream) {
    const float* query  = (const float*)d_in[0];
    const float* memory = (const float*)d_in[1];
    const float* Wq     = (const float*)d_in[2];
    const float* bq     = (const float*)d_in[3];
    const float* Wm     = (const float*)d_in[4];
    const float* v      = (const float*)d_in[5];
    float* out = (float*)d_out;
    float* weightsOut = out;                       // [BS, 1, M]
    float* wmemOut    = out + (size_t)BSZ * MDIM;  // [BS, 1, D]

    // workspace carve (all sizes 256B multiples)
    char* ws = (char*)d_ws;
    unsigned short* WmBf = (unsigned short*)ws;                 ws += (size_t)HDIM * DDIM * 2;
    float* qproj         = (float*)ws;                          ws += (size_t)BSZ * HDIM * 4;
    float* attn          = (float*)ws;                          ws += (size_t)BSZ * MDIM * 4;

    k_cvt_wm<<<(HDIM * DDIM / 4) / 256, 256, 0, stream>>>(Wm, WmBf);
    k_qproj<<<(BSZ * HDIM) / 256, 256, 0, stream>>>(query, Wq, bq, qproj);

    const size_t smem = (size_t)ROWS * STRA * 2 + (size_t)8 * ROWS * 4; // ~134KB (<320KB/WGP)
    k_scores<<<BSZ * (MDIM / ROWS), 256, smem, stream>>>(memory, WmBf, qproj, v, attn);

    k_softmax<<<BSZ, 256, 0, stream>>>(attn, weightsOut);
    k_wmem<<<BSZ * (DDIM / 256), 256, 0, stream>>>(weightsOut, memory, wmemOut);
}
